// OctopusBrain_15195594293729
// MI455X (gfx1250) — matrix operand order repaired and checked by reading, not compiled, not run on hardware
//
#include <hip/hip_runtime.h>
#include <hip/hip_bf16.h>
#include <math.h>

// ============================================================================
// OctopusBrain recurrent scan on gfx1250 (wave32 + WMMA f32_16x16x32_f16)
//
//  - Each block = 4 waves (128 threads) cooperating on ONE 16-row batch tile;
//    grid = 64 blocks covers B=1024. All T=256 steps run in-kernel with the
//    recurrent state in LDS.
//  - Per-step independent work is spread across the 4 waves:
//      optic_l || optic_r, 2 arms per wave, 2 thal N-tiles per wave.
//    Serial chain (vertical->frontal->subesoph->central) stays on wave0.
//  - Every matmul is v_wmma_f32_16x16x32_f16. A = activations (f32 LDS ->
//    v_cvt f16 per the ISA 16x32 A layout), B = time-invariant weights packed
//    once into per-lane fragment layout (f16) in d_ws (lane%16 = N column,
//    16 contiguous f16 = K rows).
//  - Intra-wave LDS producer->consumer relies on the ISA's in-order LDS
//    guarantee + a compiler motion fence; block barriers only at uniform
//    stage boundaries (WMMA always executes with EXEC all-ones).
// ============================================================================

typedef __attribute__((ext_vector_type(16))) _Float16 v16h;
typedef __attribute__((ext_vector_type(8)))  float    v8f;

#define T_STEPS 256
#define B_TOT   1024
#define HDIM    128

// region order: 0 optic_l, 1 optic_r, 2 vertical, 3 frontal,
//               4..11 arm0..7, 12 subesoph, 13 central
constexpr int RG_H[14]    = {12,12,10,10, 8,8,8,8,8,8,8,8, 7,10};
constexpr int RG_KPAD[14] = {32,32,64,64, 64,64,64,64,64,64,64,64, 96,96};
constexpr float RG_DTS[14]= {2.5f,2.5f,0.7f,0.5f, 1.8f,1.8f,1.8f,1.8f,
                             1.8f,1.8f,1.8f,1.8f, 1.5f,1.0f};
// f16-element offsets of packed fragment tiles in d_ws (tile = 512 f16)
constexpr int RG_WC[14] = {0,1024,2048,4096, 6144,8192,10240,12288,
                           14336,16384,18432,20480, 22528,25600};
constexpr int RG_WG[14] = {512,1536,3072,5120, 7168,9216,11264,13312,
                           15360,17408,19456,21504, 24064,27136};
#define WS_THAL_LIN  28672
#define WS_THAL_PROJ 45056

// LDS param-block offsets (floats). Per region: bc[H], bg[H], step[H], g[H], b[H]
constexpr int PB_RG[14] = {0,60,120,170, 220,260,300,340,380,420,460,500, 540,575};
#define PB_TLINB  625
#define PB_TPROJB 753
#define PB_TLNG   881
#define PB_TLNBB  1009
#define PB_IPW    1137
#define PB_IPB    1149
#define PB_C2F_GW 1161
#define PB_C2F_GB 1361
#define PB_C2F_PW 1381
#define PB_C2F_PB 1581
#define PB_TOT    1601

// thal concat (sorted ORDER): arm0..7, central, frontal, optic_l, optic_r,
// subesoph, vertical
constexpr int TH_START[14] = {0,8,16,24,32,40,48,56, 64,74,84,96,108,115};
constexpr int TH_LEN[14]   = {8,8,8,8,8,8,8,8, 10,10,12,12,7,10};

struct RegP {
  const float *bin, *brec, *bgin, *bgrec, *ls, *lng, *lnb;
};
struct KArgs {
  const float*    x;      // [B,T,1]
  const _Float16* wpack;  // packed fragments in d_ws
  float*          hid;    // d_out + OFF_HID, [B,T,128]
  RegP reg[14];
  const float *ipW, *ipb;                         // input_proj (1x12),(12)
  const float *c2f_gW, *c2f_gb, *c2f_pW, *c2f_pb; // (10x20),(20) x2
  const float *tlinb, *tprojb, *tlng, *tlnb;      // thal biases / ln params (128)
};

// Intra-wave fence: LDS is executed in order within a wave (ISA); this only
// stops the compiler from moving LDS ops across the producer/consumer point.
__device__ __forceinline__ void wavebar() {
  asm volatile("" ::: "memory");
  __builtin_amdgcn_wave_barrier();
  asm volatile("" ::: "memory");
}

// ---------------------------------------------------------------------------
// Weight packer: f32 [rowsA(+rowsB) x N] row-major -> f16 WMMA B-fragments.
// Fragment tile t=(nt*nKt+kt): entry [lane][e] = W[kt*32+kmap(lane>>4,e)][nt*16+(lane&15)] (0-pad)
// ---------------------------------------------------------------------------
__global__ void pack_w(const float* __restrict__ srcA, int rowsA,
                       const float* __restrict__ srcB, int rowsB,
                       int N, int kpad, _Float16* __restrict__ dst) {
  const int tile = blockIdx.x;
  const int nkt  = kpad >> 5;
  const int nt   = tile / nkt;
  const int kt   = tile % nkt;
  const int lane = threadIdx.x;
  const int hh   = lane >> 4;
  #pragma unroll
  for (int e = 0; e < 16; ++e) {
    const int K = kt * 32 + ((e < 8) ? (8 * hh + e) : (16 + 8 * hh + (e - 8)));
    int n = nt * 16 + (lane & 15);
    float v = 0.0f;
    if (n < N) {
      if (K < rowsA)                 v = srcA[K * N + n];
      else if ((K - rowsA) < rowsB)  v = srcB[(K - rowsA) * N + n];
    }
    dst[tile * 512 + lane * 16 + e] = (_Float16)v;
  }
}

// ---------------------------------------------------------------------------
// 16xKPAD (f32 LDS) x packed f16 fragments -> v8f accumulator (one N-tile)
// ---------------------------------------------------------------------------
__device__ __forceinline__ v8f gemm16(const float* __restrict__ z, int kpad,
                                      const _Float16* __restrict__ wt,
                                      int nkt, int lane) {
  v8f acc = {0.f,0.f,0.f,0.f,0.f,0.f,0.f,0.f};
  const int m = lane & 15, half = lane >> 4;
  for (int kt = 0; kt < nkt; ++kt) {
    const float* zr = z + m * kpad + kt * 32 + half * 8;
    float4 a0 = *(const float4*)(zr + 0);
    float4 a1 = *(const float4*)(zr + 4);
    float4 a2 = *(const float4*)(zr + 16);
    float4 a3 = *(const float4*)(zr + 20);
    v16h A;
    A[0]=(_Float16)a0.x;  A[1]=(_Float16)a0.y;  A[2]=(_Float16)a0.z;  A[3]=(_Float16)a0.w;
    A[4]=(_Float16)a1.x;  A[5]=(_Float16)a1.y;  A[6]=(_Float16)a1.z;  A[7]=(_Float16)a1.w;
    A[8]=(_Float16)a2.x;  A[9]=(_Float16)a2.y;  A[10]=(_Float16)a2.z; A[11]=(_Float16)a2.w;
    A[12]=(_Float16)a3.x; A[13]=(_Float16)a3.y; A[14]=(_Float16)a3.z; A[15]=(_Float16)a3.w;
    v16h Bv = *(const v16h*)(wt + kt * 512 + lane * 16);
    acc = __builtin_amdgcn_wmma_f32_16x16x32_f16(false, A, false, Bv,
                                                 (short)0, acc, false, false);
  }
  return acc;
}

// ---------------------------------------------------------------------------
// Full liquid cell (z-build + 2 GEMMs + LTC update + layernorm), wave-local.
// No block barriers inside: caller guards by wave and syncs at stage edges.
// ---------------------------------------------------------------------------
__device__ __forceinline__ void cell_full(int rg, float* __restrict__ st_h,
                                          const float* __restrict__ pbr,
                                          float* __restrict__ tmp,
                                          const _Float16* __restrict__ wpack,
                                          float* __restrict__ z,
                                          const float* const* segp, const int* segl,
                                          int nseg, int K, int lane) {
  const int kpad = RG_KPAD[rg];
  const int H    = RG_H[rg];
  // build z[16][kpad]
  for (int row = 0; row < 16; ++row) {
    float* zr = z + row * kpad;
    int off = 0;
    for (int s = 0; s < nseg; ++s) {
      const int L = segl[s];
      const float* sp = segp[s] + row * L;
      for (int j = lane; j < L; j += 32) zr[off + j] = sp[j];
      off += L;
    }
    for (int j = K + lane; j < kpad; j += 32) zr[j] = 0.f;
  }
  wavebar();
  v8f c = gemm16(z, kpad, wpack + RG_WC[rg], kpad >> 5, lane);
  v8f g = gemm16(z, kpad, wpack + RG_WG[rg], kpad >> 5, lane);
  const int n = lane & 15, half = lane >> 4;
  if (n < H) {
    const float bc = pbr[n], bg = pbr[H + n], stp = pbr[2 * H + n];
    #pragma unroll
    for (int r = 0; r < 8; ++r) {
      const int m = half * 8 + r;
      float cand = tanhf(c[r] + bc);
      float gv   = 1.f / (1.f + expf(-(g[r] + bg)));
      float hold = st_h[m * H + n];
      tmp[m * 16 + n] = hold + stp * gv * (cand - hold);
    }
  }
  wavebar();
  if (lane < 16) {
    float mu = 0.f;
    for (int j = 0; j < H; ++j) mu += tmp[lane * 16 + j];
    mu /= (float)H;
    float var = 0.f;
    for (int j = 0; j < H; ++j) { float d = tmp[lane*16+j] - mu; var += d * d; }
    var /= (float)H;
    const float inv = rsqrtf(var + 1e-5f);
    for (int j = 0; j < H; ++j)
      st_h[lane * H + j] = (tmp[lane*16+j] - mu) * inv * pbr[3*H+j] + pbr[4*H+j];
  }
  wavebar();
}

// ---------------------------------------------------------------------------
// Main recurrent scan: grid = 64 blocks x 128 threads (4 cooperating waves)
// ---------------------------------------------------------------------------
__global__ void __launch_bounds__(128, 1) octo_main(KArgs a) {
  __shared__ __align__(16) float pb[PB_TOT];
  __shared__ __align__(16) float st_optl[16*12], st_optr[16*12];
  __shared__ __align__(16) float st_vert[16*10], st_front[16*10], st_cent[16*10];
  __shared__ __align__(16) float st_arm[8*16*8], st_sub[16*7];
  __shared__ __align__(16) float zsh[16*128];        // shared z (serial stages + thal)
  __shared__ __align__(16) float zw[4][16*64];       // per-wave z (parallel stages)
  __shared__ __align__(16) float gbuf[16*128], pbuf2[16*128];
  __shared__ __align__(16) float tmpw[4][16*16];     // per-wave LTC staging
  __shared__ __align__(16) float inpbuf[16*12];
  __shared__ __align__(16) float fxbuf[16*20];

  const int tid  = threadIdx.x;
  const int lane = tid & 31;
  const int wid  = tid >> 5;
  const int b0   = blockIdx.x * 16;

  // ---- preload / precombine parameters into LDS (all 128 threads) ----
  for (int rg = 0; rg < 14; ++rg) {
    const int H = RG_H[rg], base = PB_RG[rg];
    const RegP& rp = a.reg[rg];
    for (int j = tid; j < H; j += 128) {
      pb[base + j]       = rp.bin[j]  + rp.brec[j];
      pb[base + H + j]   = rp.bgin[j] + rp.bgrec[j];
      pb[base + 2*H + j] = RG_DTS[rg] * expf(rp.ls[j]);   // DT=1
      pb[base + 3*H + j] = rp.lng[j];
      pb[base + 4*H + j] = rp.lnb[j];
    }
  }
  for (int j = tid; j < 128; j += 128) {
    pb[PB_TLINB  + j] = a.tlinb[j];
    pb[PB_TPROJB + j] = a.tprojb[j];
    pb[PB_TLNG   + j] = a.tlng[j];
    pb[PB_TLNBB  + j] = a.tlnb[j];
  }
  for (int j = tid; j < 12; j += 128) { pb[PB_IPW+j] = a.ipW[j]; pb[PB_IPB+j] = a.ipb[j]; }
  for (int j = tid; j < 200; j += 128) { pb[PB_C2F_GW+j] = a.c2f_gW[j]; pb[PB_C2F_PW+j] = a.c2f_pW[j]; }
  for (int j = tid; j < 20; j += 128)  { pb[PB_C2F_GB+j] = a.c2f_gb[j]; pb[PB_C2F_PB+j] = a.c2f_pb[j]; }

  // ---- zero initial states ----
  for (int j = tid; j < 16*12; j += 128) { st_optl[j] = 0.f; st_optr[j] = 0.f; }
  for (int j = tid; j < 16*10; j += 128) { st_vert[j] = 0.f; st_front[j] = 0.f; st_cent[j] = 0.f; }
  for (int j = tid; j < 8*16*8; j += 128) st_arm[j] = 0.f;
  for (int j = tid; j < 16*7;  j += 128) st_sub[j] = 0.f;
  __syncthreads();

  for (int t = 0; t < T_STEPS; ++t) {
    // 1) input_proj (IN=1 -> 12)
    if (tid < 16) {
      const int m = tid;
      const float xv = a.x[(size_t)(b0 + m) * T_STEPS + t];
      for (int j = 0; j < 12; ++j) inpbuf[m*12 + j] = xv * pb[PB_IPW+j] + pb[PB_IPB+j];
    }
    __syncthreads();

    // 2) optic_l (wave0) || optic_r (wave1)
    if (wid == 0) {
      const float* sp[2] = {inpbuf, st_optl}; const int sl[2] = {12,12};
      cell_full(0, st_optl, pb + PB_RG[0], tmpw[0], a.wpack, zw[0], sp, sl, 2, 24, lane);
    } else if (wid == 1) {
      const float* sp[2] = {inpbuf, st_optr}; const int sl[2] = {12,12};
      cell_full(1, st_optr, pb + PB_RG[1], tmpw[1], a.wpack, zw[1], sp, sl, 2, 24, lane);
    }
    __syncthreads();

    // 3) serial chain on wave0: vertical -> c2f axon -> frontal
    if (wid == 0) {
      { const float* sp[4] = {st_optl, st_optr, st_vert, st_vert};
        const int sl[4] = {12,12,10,10};
        cell_full(2, st_vert, pb + PB_RG[2], tmpw[0], a.wpack, zsh, sp, sl, 4, 44, lane); }
      // fx = [vert(10), cent(10)] + (t>0)*0.05*axon(c2f, cent)
      for (int idx = lane; idx < 16*20; idx += 32) {
        const int m = idx / 20, j = idx % 20;
        float base = (j < 10) ? st_vert[m*10 + j] : st_cent[m*10 + (j - 10)];
        float ax = 0.f;
        if (t > 0) {
          float p = pb[PB_C2F_PB + j], q = pb[PB_C2F_GB + j];
          for (int k = 0; k < 10; ++k) {
            const float cv = st_cent[m*10 + k];
            p += cv * pb[PB_C2F_PW + k*20 + j];
            q += cv * pb[PB_C2F_GW + k*20 + j];
          }
          ax = (1.f / (1.f + expf(-q))) * tanhf(p);
        }
        fxbuf[idx] = base + 0.05f * ax;
      }
      wavebar();
      { const float* sp[3] = {fxbuf, st_front, st_front}; const int sl[3] = {20,10,10};
        cell_full(3, st_front, pb + PB_RG[3], tmpw[0], a.wpack, zsh, sp, sl, 3, 40, lane); }
    }
    __syncthreads();

    // 4) arms: 2 per wave
    {
      #pragma unroll
      for (int q = 0; q < 2; ++q) {
        const int i = wid * 2 + q;
        float* ah = st_arm + i * 128;
        const float* sp[4] = {st_optl, st_front, ah, ah}; const int sl[4] = {12,10,8,8};
        cell_full(4 + i, ah, pb + PB_RG[4 + i], tmpw[wid], a.wpack, zw[wid],
                  sp, sl, 4, 38, lane);
      }
    }
    __syncthreads();

    // 5) wave0: subesoph -> central
    if (wid == 0) {
      { const float* sp[9]; int sl[9];
        for (int i = 0; i < 8; ++i) { sp[i] = st_arm + i * 128; sl[i] = 8; }
        sp[8] = st_sub; sl[8] = 7;
        cell_full(12, st_sub, pb + PB_RG[12], tmpw[0], a.wpack, zsh, sp, sl, 9, 71, lane); }
      { const float* sp[7] = {st_optl, st_optr, st_vert, st_front, st_sub, st_cent, st_cent};
        const int sl[7] = {12,12,10,10,7,10,10};
        cell_full(13, st_cent, pb + PB_RG[13], tmpw[0], a.wpack, zsh, sp, sl, 7, 71, lane); }
    }
    __syncthreads();

    // 6) thal: cooperative z build (ORDER concat, 125 pad 128)
    {
      const float* thp[14];
      for (int i = 0; i < 8; ++i) thp[i] = st_arm + i * 128;
      thp[8] = st_cent; thp[9] = st_front; thp[10] = st_optl; thp[11] = st_optr;
      thp[12] = st_sub; thp[13] = st_vert;
      for (int idx = tid; idx < 16 * 128; idx += 128) {
        const int row = idx >> 7, col = idx & 127;
        float v = 0.f;
        if (col < 125) {
          int s = 0;
          for (int k = 13; k >= 0; --k) { if (col >= TH_START[k]) { s = k; break; } }
          v = thp[s][row * TH_LEN[s] + (col - TH_START[s])];
        }
        zsh[idx] = v;
      }
    }
    __syncthreads();
    // 2 N-tiles per wave, both matrices
    {
      const int n = lane & 15, half = lane >> 4;
      #pragma unroll
      for (int q = 0; q < 2; ++q) {
        const int nt = wid * 2 + q;
        v8f gl = gemm16(zsh, 128, a.wpack + WS_THAL_LIN  + nt * 2048, 4, lane);
        v8f pr = gemm16(zsh, 128, a.wpack + WS_THAL_PROJ + nt * 2048, 4, lane);
        const int col = nt * 16 + n;
        #pragma unroll
        for (int r = 0; r < 8; ++r) {
          const int m = half * 8 + r;
          gbuf[m*128 + col]  = gl[r] + pb[PB_TLINB  + col];
          pbuf2[m*128 + col] = pr[r] + pb[PB_TPROJB + col];
        }
      }
    }
    __syncthreads();
    // gate = sigmoid(LN(gbuf)); out = gate * pbuf2 -> hidden[b,t,:]
    if (tid < 16) {
      const int m = tid;
      float mu = 0.f;
      for (int j = 0; j < 128; ++j) mu += gbuf[m*128 + j];
      mu *= (1.f / 128.f);
      float var = 0.f;
      for (int j = 0; j < 128; ++j) { float d = gbuf[m*128+j] - mu; var += d * d; }
      const float inv = rsqrtf(var * (1.f / 128.f) + 1e-5f);
      float* orow = a.hid + ((size_t)(b0 + m) * T_STEPS + t) * HDIM;
      for (int j = 0; j < 128; ++j) {
        const float ln = (gbuf[m*128+j] - mu) * inv * pb[PB_TLNG+j] + pb[PB_TLNBB+j];
        const float gate = 1.f / (1.f + expf(-ln));
        orow[j] = gate * pbuf2[m*128 + j];
      }
    }
    __syncthreads();
  }
}

// ---------------------------------------------------------------------------
// Post-scan heads: predictions / anomaly / surprise (1 thread per (b,t))
// ---------------------------------------------------------------------------
__global__ void heads_kernel(const float* __restrict__ hid, const float* __restrict__ x,
                             const float* __restrict__ nlg, const float* __restrict__ nlb,
                             const float* __restrict__ nW,  const float* __restrict__ nb,
                             const float* __restrict__ alg, const float* __restrict__ alb,
                             const float* __restrict__ aW,  const float* __restrict__ ab,
                             float* __restrict__ pred, float* __restrict__ anom,
                             float* __restrict__ surp) {
  const int idx = blockIdx.x * blockDim.x + threadIdx.x;
  if (idx >= B_TOT * T_STEPS) return;
  const int b = idx >> 8, t = idx & 255;
  const float* h = hid + (size_t)idx * HDIM;
  float mu = 0.f;
  for (int j = 0; j < 128; ++j) mu += h[j];
  mu *= (1.f / 128.f);
  float var = 0.f;
  for (int j = 0; j < 128; ++j) { float d = h[j] - mu; var += d * d; }
  const float inv = rsqrtf(var * (1.f / 128.f) + 1e-5f);
  float sa = 0.f;
  for (int j = 0; j < 128; ++j) sa += ((h[j] - mu) * inv * alg[j] + alb[j]) * aW[j];
  anom[idx] = 1.f / (1.f + expf(-(sa + ab[0])));
  if (t < T_STEPS - 1) {
    float sn = 0.f;
    for (int j = 0; j < 128; ++j) sn += ((h[j] - mu) * inv * nlg[j] + nlb[j]) * nW[j];
    const float p = sn + nb[0];
    pred[b * (T_STEPS - 1) + t] = p;
    const float d = p - x[(size_t)b * T_STEPS + t + 1];
    surp[b * (T_STEPS - 1) + t] = d * d;
  }
}

// proprioception head: 1 thread per batch row
__global__ void props_kernel(const float* __restrict__ hid,
                             const float* __restrict__ plg, const float* __restrict__ plb,
                             const float* __restrict__ W1, const float* __restrict__ b1,
                             const float* __restrict__ W2, const float* __restrict__ b2,
                             float* __restrict__ props) {
  const int b = blockIdx.x * blockDim.x + threadIdx.x;
  if (b >= B_TOT) return;
  const float* h = hid + ((size_t)b * T_STEPS + (T_STEPS - 1)) * HDIM;
  float mu = 0.f;
  for (int j = 0; j < 128; ++j) mu += h[j];
  mu *= (1.f / 128.f);
  float var = 0.f;
  for (int j = 0; j < 128; ++j) { float d = h[j] - mu; var += d * d; }
  const float inv = rsqrtf(var * (1.f / 128.f) + 1e-5f);
  float lnv[128];
  for (int j = 0; j < 128; ++j) lnv[j] = (h[j] - mu) * inv * plg[j] + plb[j];
  float acc0 = b2[0], acc1 = b2[1];
  for (int o = 0; o < 64; ++o) {
    float s = b1[o];
    for (int j = 0; j < 128; ++j) s += lnv[j] * W1[j * 64 + o];
    const float gz = 0.5f * s * (1.f + erff(s * 0.70710678118654752f));
    acc0 += gz * W2[o * 2 + 0];
    acc1 += gz * W2[o * 2 + 1];
  }
  props[b * 2 + 0] = acc0;
  props[b * 2 + 1] = acc1;
}

// ---------------------------------------------------------------------------
// Host: d_in[0]=x, then params tree flattened with sorted dict keys:
//  anom_lin(W,b)=1,2  anom_ln(b,g)=3,4  arms[i]: base 5+11i with leaves
//  {ginW,ginb,grecW,grecb,inW,inb,log_step,recW,recb,norm.b,norm.g}
//  c2f(gateW,gateb,projW,projb)=93..96  central=97  frontal=108
//  input_proj(W,b)=119,120  next_lin=121,122  next_ln(b,g)=123,124
//  optic_l=125  optic_r=136  prop_l1=147,148  prop_l2=149,150
//  prop_ln(b,g)=151,152  subesoph=153  thal_lin=164,165  thal_ln(b,g)=166,167
//  thal_proj=168,169  vertical=170
// ---------------------------------------------------------------------------
extern "C" void kernel_launch(void* const* d_in, const int* in_sizes, int n_in,
                              void* d_out, int out_size, void* d_ws, size_t ws_size,
                              hipStream_t stream) {
  (void)in_sizes; (void)n_in; (void)out_size; (void)ws_size;
  auto F = [&](int i) { return (const float*)d_in[i]; };
  _Float16* ws = (_Float16*)d_ws;
  float* out = (float*)d_out;

  const size_t OFF_PRED  = 0;
  const size_t OFF_HID   = (size_t)B_TOT * (T_STEPS - 1);
  const size_t OFF_ANOM  = OFF_HID  + (size_t)B_TOT * T_STEPS * HDIM;
  const size_t OFF_SURP  = OFF_ANOM + (size_t)B_TOT * T_STEPS;
  const size_t OFF_PROPS = OFF_SURP + (size_t)B_TOT * (T_STEPS - 1);

  // region param base index in flat input list (kernel region order)
  static const int rbase[14] = {125,136,170,108, 5,16,27,38,49,60,71,82, 153,97};
  static const int rE[14]    = {12,12,34,30, 30,30,30,30,30,30,30,30, 64,61};
  static const int rH[14]    = {12,12,10,10, 8,8,8,8,8,8,8,8, 7,10};

  // ---- pack weights to f16 fragments (stacked [W_in;W_rec] / [W_gin;W_grec])
  for (int rg = 0; rg < 14; ++rg) {
    const int grid = RG_KPAD[rg] / 32;   // H<=16 -> single N tile
    pack_w<<<grid, 32, 0, stream>>>(F(rbase[rg] + 4), rE[rg],
                                    F(rbase[rg] + 7), rH[rg],
                                    rH[rg], RG_KPAD[rg], ws + RG_WC[rg]);
    pack_w<<<grid, 32, 0, stream>>>(F(rbase[rg] + 0), rE[rg],
                                    F(rbase[rg] + 2), rH[rg],
                                    rH[rg], RG_KPAD[rg], ws + RG_WG[rg]);
  }
  pack_w<<<32, 32, 0, stream>>>(F(164), 125, nullptr, 0, 128, 128, ws + WS_THAL_LIN);
  pack_w<<<32, 32, 0, stream>>>(F(168), 125, nullptr, 0, 128, 128, ws + WS_THAL_PROJ);

  // ---- main recurrent scan ----
  KArgs a;
  a.x = F(0);
  a.wpack = ws;
  a.hid = out + OFF_HID;
  for (int rg = 0; rg < 14; ++rg) {
    const int b = rbase[rg];
    a.reg[rg].bin   = F(b + 5);   // in.b
    a.reg[rg].brec  = F(b + 8);   // rec.b
    a.reg[rg].bgin  = F(b + 1);   // gin.b
    a.reg[rg].bgrec = F(b + 3);   // grec.b
    a.reg[rg].ls    = F(b + 6);   // log_step
    a.reg[rg].lng   = F(b + 10);  // norm.g
    a.reg[rg].lnb   = F(b + 9);   // norm.b
  }
  a.ipW = F(119);  a.ipb = F(120);
  a.c2f_gW = F(93); a.c2f_gb = F(94); a.c2f_pW = F(95); a.c2f_pb = F(96);
  a.tlinb = F(165); a.tprojb = F(169); a.tlng = F(167); a.tlnb = F(166);
  octo_main<<<B_TOT / 16, 128, 0, stream>>>(a);

  // ---- post-scan heads ----
  heads_kernel<<<(B_TOT * T_STEPS) / 256, 256, 0, stream>>>(
      out + OFF_HID, F(0),
      F(124), F(123), F(121), F(122),   // next_ln.g, next_ln.b, next_lin.W, next_lin.b
      F(4),   F(3),   F(1),   F(2),     // anom_ln.g, anom_ln.b, anom_lin.W, anom_lin.b
      out + OFF_PRED, out + OFF_ANOM, out + OFF_SURP);

  props_kernel<<<(B_TOT + 127) / 128, 128, 0, stream>>>(
      out + OFF_HID, F(152), F(151), F(147), F(148), F(149), F(150),
      out + OFF_PROPS);
}
